// pointnet2encoder_18760417149509
// MI455X (gfx1250) — matrix operand order repaired and checked by reading, not compiled, not run on hardware
//
#include <hip/hip_runtime.h>
#include <hip/hip_bf16.h>
#include <cstdint>
#include <cstddef>

typedef __attribute__((ext_vector_type(16))) _Float16 v16h;
typedef __attribute__((ext_vector_type(8)))  _Float16 v8h;
typedef __attribute__((ext_vector_type(8)))  float    v8f;

// ---------------------------------------------------------------------------
// Layout transposes
// ---------------------------------------------------------------------------
__global__ void k_transpose_in(const float* __restrict__ xyz, float* __restrict__ out,
                               int B, int N) {
  int i = blockIdx.x * blockDim.x + threadIdx.x;
  int total = B * 3 * N;
  if (i >= total) return;
  int n = i % N;
  int c = (i / N) % 3;
  int b = i / (3 * N);
  out[((size_t)b * N + n) * 3 + c] = xyz[i];
}

__global__ void k_transpose_out_h(const _Float16* __restrict__ in, float* __restrict__ out,
                                  int B, int N, int C) {
  int i = blockIdx.x * blockDim.x + threadIdx.x;
  int total = B * C * N;
  if (i >= total) return;
  int n = i % N;
  int c = (i / N) % C;
  int b = i / (C * N);
  out[i] = (float)in[((size_t)b * N + n) * C + c];
}

// ---------------------------------------------------------------------------
// Farthest point sampling: one block per batch, sequential over npoint iters.
// LDS holds the running min-distance array (N <= 8192 -> 32KB, fine in 320KB).
// ---------------------------------------------------------------------------
__global__ void k_fps(const float* __restrict__ xyz, int N, int npoint,
                      int* __restrict__ out) {
  __shared__ float sd[8192];
  __shared__ float rmax[256];
  __shared__ int   rarg[256];
  __shared__ int   s_far;
  int b = blockIdx.x;
  int t = threadIdx.x;
  const float* X = xyz + (size_t)b * N * 3;
  for (int i = t; i < N; i += 256) sd[i] = 1e10f;
  if (t == 0) s_far = 0;
  __syncthreads();
  for (int it = 0; it < npoint; ++it) {
    int far = s_far;
    if (t == 0) out[b * npoint + it] = far;
    float cx = X[far * 3 + 0], cy = X[far * 3 + 1], cz = X[far * 3 + 2];
    float best = -1.0f; int barg = 0;
    for (int i = t; i < N; i += 256) {
      float dx = X[i * 3 + 0] - cx;
      float dy = X[i * 3 + 1] - cy;
      float dz = X[i * 3 + 2] - cz;
      float d = dx * dx + dy * dy + dz * dz;
      float nd = fminf(sd[i], d);
      sd[i] = nd;
      if (nd > best) { best = nd; barg = i; }
    }
    rmax[t] = best; rarg[t] = barg;
    __syncthreads();
    for (int s = 128; s > 0; s >>= 1) {
      if (t < s && rmax[t + s] > rmax[t]) { rmax[t] = rmax[t + s]; rarg[t] = rarg[t + s]; }
      __syncthreads();
    }
    if (t == 0) s_far = rarg[0];
    __syncthreads();
  }
}

__global__ void k_gather_xyz(const float* __restrict__ xyz, const int* __restrict__ idx,
                             int S, int N, float* __restrict__ out) {
  int i = blockIdx.x * blockDim.x + threadIdx.x;  // over B*S
  if (i >= 2 * S) return;
  int b = i / S, s = i % S;
  int n = idx[b * S + s];
  const float* src = xyz + ((size_t)b * N + n) * 3;
  float* dst = out + (size_t)i * 3;
  dst[0] = src[0]; dst[1] = src[1]; dst[2] = src[2];
}

// ---------------------------------------------------------------------------
// Ball query: first `nsample` indices (ascending) within radius, pad w/ first.
// ---------------------------------------------------------------------------
__global__ void k_ballq(const float* __restrict__ xyz, const float* __restrict__ nxyz,
                        int N, int S, int nsample, float r2, int* __restrict__ idx) {
  int i = blockIdx.x * blockDim.x + threadIdx.x;  // over B*S
  if (i >= 2 * S) return;
  int b = i / S;
  const float* X = xyz + (size_t)b * N * 3;
  float qx = nxyz[(size_t)i * 3 + 0];
  float qy = nxyz[(size_t)i * 3 + 1];
  float qz = nxyz[(size_t)i * 3 + 2];
  int* o = idx + (size_t)i * nsample;
  int cnt = 0, first = 0;
  for (int n = 0; n < N && cnt < nsample; ++n) {
    float dx = X[n * 3 + 0] - qx;
    float dy = X[n * 3 + 1] - qy;
    float dz = X[n * 3 + 2] - qz;
    float d = dx * dx + dy * dy + dz * dz;
    if (d <= r2) { if (cnt == 0) first = n; o[cnt++] = n; }
  }
  for (; cnt < nsample; ++cnt) o[cnt] = first;
}

// grouped row (f16, stride Kp, zero-padded) = concat(xyz[idx]-new_xyz, points[idx])
__global__ void k_group_h(const float* __restrict__ xyz, const float* __restrict__ nxyz,
                          const float* __restrict__ pts, const int* __restrict__ idx,
                          int N, int S, int K, int Cp, int Kp,
                          _Float16* __restrict__ out) {
  int i = blockIdx.x * blockDim.x + threadIdx.x;  // over B*S*K rows
  if (i >= 2 * S * K) return;
  int bs = i / K;
  int b = bs / S;
  int n = idx[i];
  _Float16* o = out + (size_t)i * Kp;
  const float* src = xyz + ((size_t)b * N + n) * 3;
  const float* q   = nxyz + (size_t)bs * 3;
  o[0] = (_Float16)(src[0] - q[0]);
  o[1] = (_Float16)(src[1] - q[1]);
  o[2] = (_Float16)(src[2] - q[2]);
  const float* ps = pts + ((size_t)b * N + n) * Cp;
  for (int c = 0; c < Cp; ++c) o[3 + c] = (_Float16)ps[c];
  for (int c = 3 + Cp; c < Kp; ++c) o[c] = (_Float16)0.0f;
}

// ---------------------------------------------------------------------------
// Weight pad-convert: W (OxK f32) -> Wh (OxKp f16), zero-padded along K.
// ---------------------------------------------------------------------------
__global__ void k_convert_w(const float* __restrict__ W, _Float16* __restrict__ Wh,
                            int O, int K, int Kp) {
  int i = blockIdx.x * blockDim.x + threadIdx.x;
  if (i >= O * Kp) return;
  int o = i / Kp, k = i % Kp;
  Wh[i] = (k < K) ? (_Float16)W[(size_t)o * K + k] : (_Float16)0.0f;
}

// ---------------------------------------------------------------------------
// WMMA GEMM: C(MxO) = A(MxKp f16) * Wh(OxKp f16)^T + bias, f32 accumulate.
// Specialized on NT = O-tiles per wave (2 or 4): zero guards in the hot loop,
// one A fragment reused by NT back-to-back v_wmma on independent accumulators.
// Operands pre-padded (Kp % 32 == 0) -> contiguous b128 loads (k = 16*hi + s, same for A and B).
// ---------------------------------------------------------------------------
template <int NT>
__global__ void k_wmma_gemm_t(const _Float16* __restrict__ A,
                              const _Float16* __restrict__ Wh,
                              const float* __restrict__ bias, float* __restrict__ C,
                              int M, int Kp, int O) {
  int tilesM = M >> 4;
  int groupsO = (O >> 4) / NT;               // exact by dispatch choice
  int gw = (blockIdx.x * blockDim.x + threadIdx.x) >> 5;
  if (gw >= tilesM * groupsO) return;        // wave-uniform -> EXEC all-ones inside
  int tm = gw % tilesM;
  int tg = gw / tilesM;
  int lane = threadIdx.x & 31;
  int l16 = lane & 15;
  int hi  = lane >> 4;

  // A fragment: lane l16 = row; halves 0..15 at K=k0+hi*16 (contiguous)
  const _Float16* arow = A + (size_t)(tm * 16 + l16) * Kp + hi * 16;
  // B fragment: lane l16 = out-channel; halves 0..15 at K=k0+hi*16 (contiguous)
  const _Float16* wr[NT];
#pragma unroll
  for (int j = 0; j < NT; ++j)
    wr[j] = Wh + (size_t)(tg * (NT * 16) + j * 16 + l16) * Kp + hi * 16;

  v8f acc[NT];
#pragma unroll
  for (int j = 0; j < NT; ++j) acc[j] = (v8f){};

  for (int k0 = 0; k0 < Kp; k0 += 32) {
    __builtin_prefetch(arow + k0 + 256, 0, 1);
    v8h al = *(const v8h*)(arow + k0);
    v8h ah = *(const v8h*)(arow + k0 + 8);
    v16h a = __builtin_shufflevector(al, ah, 0, 1, 2, 3, 4, 5, 6, 7,
                                             8, 9, 10, 11, 12, 13, 14, 15);
#pragma unroll
    for (int j = 0; j < NT; ++j) {
      v16h b = *(const v16h*)(wr[j] + k0);
      acc[j] = __builtin_amdgcn_wmma_f32_16x16x32_f16(false, a, false, b,
                                                      (short)0, acc[j], false, false);
    }
  }

  // C/D layout: VGPR r -> M = r + 8*hi, N = l16
  int mbase = tm * 16 + (hi ? 8 : 0);
#pragma unroll
  for (int r = 0; r < 8; ++r) {
    float* crow = C + (size_t)(mbase + r) * O + l16;
#pragma unroll
    for (int j = 0; j < NT; ++j) {
      int n = tg * (NT * 16) + j * 16;
      crow[n] = acc[j][r] + bias[n + l16];
    }
  }
}

// ---------------------------------------------------------------------------
// BatchNorm batch-statistics (population var) + fused normalize/ReLU -> f16.
// ---------------------------------------------------------------------------
__global__ void k_bn_stats(const float* __restrict__ X, int M, int O,
                           float* __restrict__ mean, float* __restrict__ inv) {
  __shared__ float s1[256], s2[256];
  int o = blockIdx.x;
  int t = threadIdx.x;
  float a = 0.0f, b = 0.0f;
  for (int m = t; m < M; m += 256) {
    float x = X[(size_t)m * O + o];
    a += x; b += x * x;
  }
  s1[t] = a; s2[t] = b;
  __syncthreads();
  for (int s = 128; s > 0; s >>= 1) {
    if (t < s) { s1[t] += s1[t + s]; s2[t] += s2[t + s]; }
    __syncthreads();
  }
  if (t == 0) {
    float mu = s1[0] / (float)M;
    float var = s2[0] / (float)M - mu * mu;
    mean[o] = mu;
    inv[o] = rsqrtf(fmaxf(var, 0.0f) + 1e-5f);
  }
}

__global__ void k_bn_relu_h(const float* __restrict__ X, _Float16* __restrict__ Y,
                            int M, int O,
                            const float* __restrict__ mean, const float* __restrict__ inv,
                            const float* __restrict__ g, const float* __restrict__ bt) {
  size_t i = (size_t)blockIdx.x * blockDim.x + threadIdx.x;
  size_t total = (size_t)M * O;
  if (i >= total) return;
  int o = (int)(i % O);
  float x = (X[i] - mean[o]) * inv[o] * g[o] + bt[o];
  Y[i] = (_Float16)fmaxf(x, 0.0f);
}

__global__ void k_maxpool_h(const _Float16* __restrict__ X, int BS, int K, int O,
                            float* __restrict__ out) {
  int i = blockIdx.x * blockDim.x + threadIdx.x;  // over BS*O
  if (i >= BS * O) return;
  int bs = i / O, o = i % O;
  float m = -3.4e38f;
  for (int k = 0; k < K; ++k)
    m = fmaxf(m, (float)X[((size_t)bs * K + k) * O + o]);
  out[i] = m;
}

__global__ void k_copy_h2f(const _Float16* __restrict__ src, float* __restrict__ dst,
                           long n) {
  long i = (long)blockIdx.x * blockDim.x + threadIdx.x;
  if (i >= n) return;
  dst[i] = (float)src[i];
}

// ---------------------------------------------------------------------------
// 3-NN + inverse-distance weights, then interpolation with optional concat.
// ---------------------------------------------------------------------------
__global__ void k_knn3(const float* __restrict__ x1, const float* __restrict__ x2,
                       int N1, int N2, int* __restrict__ idx, float* __restrict__ w) {
  int i = blockIdx.x * blockDim.x + threadIdx.x;  // over B*N1
  if (i >= 2 * N1) return;
  int b = i / N1;
  float qx = x1[(size_t)i * 3 + 0];
  float qy = x1[(size_t)i * 3 + 1];
  float qz = x1[(size_t)i * 3 + 2];
  const float* X = x2 + (size_t)b * N2 * 3;
  float d0 = 3.4e38f, d1 = 3.4e38f, d2 = 3.4e38f;
  int i0 = 0, i1 = 0, i2 = 0;
  for (int m = 0; m < N2; ++m) {
    float dx = X[m * 3 + 0] - qx;
    float dy = X[m * 3 + 1] - qy;
    float dz = X[m * 3 + 2] - qz;
    float d = dx * dx + dy * dy + dz * dz;
    if (d < d0)      { d2 = d1; i2 = i1; d1 = d0; i1 = i0; d0 = d; i0 = m; }
    else if (d < d1) { d2 = d1; i2 = i1; d1 = d;  i1 = m; }
    else if (d < d2) { d2 = d;  i2 = m; }
  }
  float r0 = 1.0f / (fmaxf(d0, 0.0f) + 1e-8f);
  float r1 = 1.0f / (fmaxf(d1, 0.0f) + 1e-8f);
  float r2 = 1.0f / (fmaxf(d2, 0.0f) + 1e-8f);
  float s = r0 + r1 + r2;
  idx[(size_t)i * 3 + 0] = i0;
  idx[(size_t)i * 3 + 1] = i1;
  idx[(size_t)i * 3 + 2] = i2;
  w[(size_t)i * 3 + 0] = r0 / s;
  w[(size_t)i * 3 + 1] = r1 / s;
  w[(size_t)i * 3 + 2] = r2 / s;
}

__global__ void k_interp_concat_h(const float* __restrict__ p1, int C1,
                                  const float* __restrict__ p2, int C2,
                                  const int* __restrict__ idx, const float* __restrict__ w,
                                  int N1, int N2, _Float16* __restrict__ X) {
  int C = C1 + C2;
  size_t i = (size_t)blockIdx.x * blockDim.x + threadIdx.x;  // over B*N1*C
  if (i >= (size_t)2 * N1 * C) return;
  int c = (int)(i % C);
  size_t bn = i / C;
  int b = (int)(bn / N1);
  if (c < C1) {
    X[i] = (_Float16)p1[bn * C1 + c];
  } else {
    int c2 = c - C1;
    const int*   id = idx + bn * 3;
    const float* ww = w + bn * 3;
    float v = ww[0] * p2[((size_t)b * N2 + id[0]) * C2 + c2]
            + ww[1] * p2[((size_t)b * N2 + id[1]) * C2 + c2]
            + ww[2] * p2[((size_t)b * N2 + id[2]) * C2 + c2];
    X[i] = (_Float16)v;
  }
}

// ---------------------------------------------------------------------------
// Host orchestration
// ---------------------------------------------------------------------------
struct ConvP { const float *W, *b, *g, *bt; int cin, cout; };

extern "C" void kernel_launch(void* const* d_in, const int* in_sizes, int n_in,
                              void* d_out, int out_size, void* d_ws, size_t ws_size,
                              hipStream_t stream) {
  (void)in_sizes; (void)n_in; (void)out_size; (void)ws_size;
  const float* xyz_in = (const float*)d_in[0];
  int p = 1;
  auto nextp = [&]() { return (const float*)d_in[p++]; };

  const int sa_cin[4] = {6, 67, 131, 259};
  const int sa_mlp[4][3] = {{32,32,64},{64,64,128},{128,128,256},{256,256,512}};
  const int fp_cin[4] = {768, 384, 320, 128};
  const int fp_mlp[4][3] = {{256,256,0},{256,256,0},{256,128,0},{128,128,128}};
  const int fp_nc[4] = {2, 2, 2, 3};

  ConvP sa[4][3], fpc[4][3];
  for (int s = 0; s < 4; ++s) {
    int c = sa_cin[s];
    for (int j = 0; j < 3; ++j) {
      ConvP cv; cv.cin = c; cv.cout = sa_mlp[s][j];
      cv.W = nextp(); cv.b = nextp(); cv.g = nextp(); cv.bt = nextp();
      sa[s][j] = cv; c = cv.cout;
    }
  }
  for (int s = 0; s < 4; ++s) {
    int c = fp_cin[s];
    for (int j = 0; j < fp_nc[s]; ++j) {
      ConvP cv; cv.cin = c; cv.cout = fp_mlp[s][j];
      cv.W = nextp(); cv.b = nextp(); cv.g = nextp(); cv.bt = nextp();
      fpc[s][j] = cv; c = cv.cout;
    }
  }

  // ----- workspace bump allocator -----
  char* wsp = (char*)d_ws;
  auto alloc = [&](size_t bytes) -> void* {
    void* r = wsp; wsp += (bytes + 255) & ~(size_t)255; return r;
  };
  const int B = 2;
  const int Ns[5] = {8192, 2048, 512, 128, 32};
  const int Cf[5] = {3, 64, 128, 256, 512};
  float* lx[5];
  for (int i = 0; i < 5; ++i) lx[i] = (float*)alloc((size_t)B * Ns[i] * 3 * 4);
  float* lf[5];
  lf[0] = lx[0];
  for (int i = 1; i < 5; ++i) lf[i] = (float*)alloc((size_t)B * Ns[i] * Cf[i] * 4);
  float* l3p = (float*)alloc((size_t)B * 128 * 256 * 4);
  float* l2p = (float*)alloc((size_t)B * 512 * 256 * 4);
  float* l1p = (float*)alloc((size_t)B * 2048 * 128 * 4);
  _Float16* hA   = (_Float16*)alloc((size_t)8388608 * 2);
  _Float16* hB   = (_Float16*)alloc((size_t)8388608 * 2);
  float*    gemC = (float*)alloc((size_t)8388608 * 4);
  _Float16* wbuf = (_Float16*)alloc((size_t)262144 * 2);
  float* mean = (float*)alloc(512 * 4);
  float* invs = (float*)alloc(512 * 4);
  int* fps_idx = (int*)alloc((size_t)B * 2048 * 4);
  int* ball    = (int*)alloc((size_t)B * 2048 * 32 * 4);
  int* knni    = (int*)alloc((size_t)B * 8192 * 3 * 4);
  float* knnw  = (float*)alloc((size_t)B * 8192 * 3 * 4);

  auto cdiv = [](long a, long b) { return (unsigned)((a + b - 1) / b); };
  auto pad32 = [](int k) { return (k + 31) & ~31; };

  auto run_mlp = [&](_Float16*& x, _Float16*& y, long M, ConvP* convs, int nconv) {
    for (int j = 0; j < nconv; ++j) {
      int K = convs[j].cin, Kp = pad32(K), O = convs[j].cout;
      k_convert_w<<<cdiv((long)O * Kp, 256), 256, 0, stream>>>(convs[j].W, wbuf, O, K, Kp);
      int tilesO = O / 16;
      if ((tilesO & 3) == 0) {
        long waves = (M / 16) * (tilesO / 4);
        k_wmma_gemm_t<4><<<cdiv(waves, 4), 128, 0, stream>>>(x, wbuf, convs[j].b, gemC,
                                                             (int)M, Kp, O);
      } else {
        long waves = (M / 16) * (tilesO / 2);
        k_wmma_gemm_t<2><<<cdiv(waves, 4), 128, 0, stream>>>(x, wbuf, convs[j].b, gemC,
                                                             (int)M, Kp, O);
      }
      k_bn_stats<<<O, 256, 0, stream>>>(gemC, (int)M, O, mean, invs);
      k_bn_relu_h<<<cdiv(M * O, 256), 256, 0, stream>>>(gemC, y, (int)M, O, mean, invs,
                                                        convs[j].g, convs[j].bt);
      _Float16* t = x; x = y; y = t;
    }
  };

  // ----- l0_xyz = transpose(xyz) -----
  k_transpose_in<<<cdiv((long)B * 3 * 8192, 256), 256, 0, stream>>>(xyz_in, lx[0], B, 8192);

  // ----- set-abstraction stages -----
  const float r2s[4] = {0.01f, 0.04f, 0.16f, 0.64f};
  const int npts[4] = {2048, 512, 128, 32};
  for (int s = 0; s < 4; ++s) {
    int Nin = Ns[s], S = npts[s], Cp = Cf[s];
    int Kp = pad32(3 + Cp);
    k_fps<<<B, 256, 0, stream>>>(lx[s], Nin, S, fps_idx);
    k_gather_xyz<<<cdiv((long)B * S, 256), 256, 0, stream>>>(lx[s], fps_idx, S, Nin, lx[s + 1]);
    k_ballq<<<cdiv((long)B * S, 128), 128, 0, stream>>>(lx[s], lx[s + 1], Nin, S, 32,
                                                        r2s[s], ball);
    long Mrows = (long)B * S * 32;
    k_group_h<<<cdiv(Mrows, 256), 256, 0, stream>>>(lx[s], lx[s + 1], lf[s], ball,
                                                    Nin, S, 32, Cp, Kp, hA);
    _Float16* xb = hA; _Float16* yb = hB;
    run_mlp(xb, yb, Mrows, sa[s], 3);
    int O = sa[s][2].cout;
    k_maxpool_h<<<cdiv((long)B * S * O, 256), 256, 0, stream>>>(xb, B * S, 32, O, lf[s + 1]);
  }

  // ----- feature-propagation stages -----
  struct FPDesc {
    const float* x1; const float* x2; int N1, N2;
    const float* p1; int C1; const float* p2; int C2;
    ConvP* cv; int nc; float* out;
  };
  FPDesc fpd[4] = {
    { lx[3], lx[4],  128,   32, lf[3], 256, lf[4], 512, fpc[0], 2, l3p },
    { lx[2], lx[3],  512,  128, lf[2], 128, l3p,   256, fpc[1], 2, l2p },
    { lx[1], lx[2], 2048,  512, lf[1],  64, l2p,   256, fpc[2], 2, l1p },
    { lx[0], lx[1], 8192, 2048, nullptr, 0, l1p,   128, fpc[3], 3, nullptr },
  };
  for (int s = 0; s < 4; ++s) {
    FPDesc& d = fpd[s];
    k_knn3<<<cdiv((long)B * d.N1, 128), 128, 0, stream>>>(d.x1, d.x2, d.N1, d.N2,
                                                          knni, knnw);
    int C = d.C1 + d.C2;  // all multiples of 32
    k_interp_concat_h<<<cdiv((long)B * d.N1 * C, 256), 256, 0, stream>>>(
        d.p1, d.C1, d.p2, d.C2, knni, knnw, d.N1, d.N2, hA);
    _Float16* xb = hA; _Float16* yb = hB;
    run_mlp(xb, yb, (long)B * d.N1, d.cv, d.nc);
    int O = d.cv[d.nc - 1].cout;
    if (d.out) {
      k_copy_h2f<<<cdiv((long)B * d.N1 * O, 256), 256, 0, stream>>>(
          xb, d.out, (long)B * d.N1 * O);
    } else {
      k_transpose_out_h<<<cdiv((long)B * O * d.N1, 256), 256, 0, stream>>>(
          xb, (float*)d_out, B, d.N1, O);
    }
  }
}
